// GraphSage_3693671875294
// MI455X (gfx1250) — matrix operand order repaired and checked by reading, not compiled, not run on hardware
//
#include <hip/hip_runtime.h>
#include <hip/hip_bf16.h>

// ---------------------------------------------------------------------------
// GraphSAGE (2x SAGEConv + link head) for MI455X / gfx1250.
//
// Strategy (see analysis): edge-parallel atomic scatter-mean (memory-bound,
// ~410MB random reads -> ~20-40us floor at 23.3TB/s), then WMMA GEMMs in
// split-precision bf16 (hi/lo, 3x v_wmma_f32_16x16x32_bf16 per 16x16x32 tile,
// f32 accumulate) which retains ~fp32 accuracy at matrix-core rates.
// A and B operands are pre-packed into the exact CDNA5 per-lane fragment
// layouts so the GEMM inner loop is only coalesced b128 loads + WMMA.
// ---------------------------------------------------------------------------

typedef __attribute__((ext_vector_type(16))) __bf16 bf16x16;
typedef __attribute__((ext_vector_type(8)))  float  f32x8;

union Frag {           // one WMMA operand fragment: 8 VGPRs per lane
    bf16x16  v;
    unsigned u[8];
};

// --- bf16 split-precision helpers (round-to-nearest-even) ------------------
__device__ __forceinline__ unsigned bf16h(float f) {
    unsigned u = __float_as_uint(f);
    return (u + 0x7fffu + ((u >> 16) & 1u)) >> 16;
}
__device__ __forceinline__ float bf16f(unsigned h) {
    return __uint_as_float(h << 16);
}
// Split two consecutive-K f32 values into packed hi/lo bf16 dwords
// (low 16 bits = even K, high 16 bits = odd K, matching VGPR halfword order).
__device__ __forceinline__ void split2(float a, float b, unsigned& hi, unsigned& lo) {
    unsigned ha = bf16h(a), hb = bf16h(b);
    unsigned la = bf16h(a - bf16f(ha));
    unsigned lb = bf16h(b - bf16f(hb));
    hi = ha | (hb << 16);
    lo = la | (lb << 16);
}

// ---------------------------------------------------------------------------
// Zero fill (graph-capture safe replacement for memset)
// ---------------------------------------------------------------------------
__global__ void zero_f4(float4* __restrict__ p, int n4) {
    int t = blockIdx.x * blockDim.x + threadIdx.x;
    if (t < n4) p[t] = make_float4(0.f, 0.f, 0.f, 0.f);
}

// ---------------------------------------------------------------------------
// Edge-parallel scatter: agg[dst] += x[src], cnt[dst] += 1.
// 64 threads (float4 chunks) per edge; index loads broadcast from cache.
// ---------------------------------------------------------------------------
__global__ void scatter_add(const float* __restrict__ x,
                            const int* __restrict__ src,
                            const int* __restrict__ dst,
                            float* __restrict__ agg,
                            float* __restrict__ cnt, int E) {
    int t = blockIdx.x * blockDim.x + threadIdx.x;
    if (t >= E * 64) return;
    int e = t >> 6, c = t & 63;
    int s = src[e], d = dst[e];
    const float4 v = *(const float4*)(x + (size_t)s * 256 + c * 4);
    float* a = agg + (size_t)d * 256 + c * 4;
    atomicAdd(a + 0, v.x);
    atomicAdd(a + 1, v.y);
    atomicAdd(a + 2, v.z);
    atomicAdd(a + 3, v.w);
    if (c == 0) atomicAdd(cnt + d, 1.0f);
}

// ---------------------------------------------------------------------------
// Finalize mean and emit the concatenated A operand [mean | self] (M x 512)
// directly in packed hi/lo bf16 WMMA A-fragment layout.
//
// Fragment tile = (mtile, ktile) covering 16 rows x 32 K, stored as 512
// dwords: dword index = chunk*128 + lane*4 + (j&3); hi dwords at chunks 0..1
// (j=0..7), lo dwords at chunks 2..3. Lane mapping (ISA 16-bit A 16x32):
//   lane = mm + 16*kh,  kh = (kk>>3)&1
//   j    = (kk&16 ? 4 : 0) + ((kk&7)>>1),  halfword = kk&1
// ---------------------------------------------------------------------------
__global__ void finalize_pack_A(const float* __restrict__ agg,
                                const float* __restrict__ cnt,
                                const float* __restrict__ self,
                                unsigned* __restrict__ Ap, int n) {
    int t = blockIdx.x * blockDim.x + threadIdx.x;
    if (t >= n * 256) return;
    int m  = t >> 8;
    int kp = t & 255;          // even-K pair index, k = 2*kp in [0,512)
    int k  = kp * 2;
    float a, b;
    if (k < 256) {             // mean half
        float ic = 1.0f / fmaxf(cnt[m], 1.0f);
        const float2 v = *(const float2*)(agg + (size_t)m * 256 + k);
        a = v.x * ic; b = v.y * ic;
    } else {                   // self half
        const float2 v = *(const float2*)(self + (size_t)m * 256 + (k - 256));
        a = v.x; b = v.y;
    }
    unsigned hi, lo;
    split2(a, b, hi, lo);
    int mtile = m >> 4, mm = m & 15;
    int ktile = k >> 5, kk = k & 31;
    int kh   = (kk >> 3) & 1;
    int j    = ((kk & 16) ? 4 : 0) + ((kk & 7) >> 1);
    int lane = mm + 16 * kh;
    unsigned* tile = Ap + (((size_t)mtile * 16 + ktile) << 9);
    tile[(j >> 2) * 128 + lane * 4 + (j & 3)]       = hi;
    tile[(2 + (j >> 2)) * 128 + lane * 4 + (j & 3)] = lo;
}

// ---------------------------------------------------------------------------
// Pack stacked weights [Wl; Wr] (512 x 256) into hi/lo bf16 WMMA B-fragment
// tiles (32 K x 16 N each). Lane mapping (ISA 16-bit B 32x16):
//   lane = nn + 16*kh2, kh2 = (kk>>3)&1, j = (kk&16 ? 4 : 0) + ((kk&7)>>1), halfword = kk&1
// Same 512-dword-per-tile chunk layout as A.
// ---------------------------------------------------------------------------
__global__ void pack_B(const float* __restrict__ Wl,
                       const float* __restrict__ Wr,
                       unsigned* __restrict__ Bp) {
    int t = blockIdx.x * blockDim.x + threadIdx.x;   // 16*16*32*8 = 65536
    if (t >= 65536) return;
    int j     = t & 7;
    int lane  = (t >> 3) & 31;
    int ntile = (t >> 8) & 15;
    int kt    = t >> 12;
    int kh2 = lane >> 4, nn = lane & 15;
    int n  = ntile * 16 + nn;
    int k0 = kt * 32 + ((j & 4) ? 16 : 0) + kh2 * 8 + 2 * (j & 3);             // even, pairs never straddle 256
    float a = (k0 < 256) ? Wl[k0 * 256 + n] : Wr[(k0 - 256) * 256 + n];
    float b = (k0 + 1 < 256) ? Wl[(k0 + 1) * 256 + n] : Wr[(k0 + 1 - 256) * 256 + n];
    unsigned hi, lo;
    split2(a, b, hi, lo);
    unsigned* tile = Bp + (((size_t)kt * 16 + ntile) << 9);
    tile[(j >> 2) * 128 + lane * 4 + (j & 3)]       = hi;
    tile[(2 + (j >> 2)) * 128 + lane * 4 + (j & 3)] = lo;
}

// ---------------------------------------------------------------------------
// GEMM: Out(M x 256) = relu( A(M x 512) @ B(512 x 256) + bias ), operands
// pre-packed. Block = 4 waves; wave w owns N columns [w*64, w*64+64) of one
// 16-row strip (4 accumulators). K loop: 16 steps of 32, 3 WMMAs per tile
// (split-precision bf16). No divergence -> EXEC all-ones for WMMA.
// ---------------------------------------------------------------------------
__global__ void __launch_bounds__(128)
gemm_bias_relu(const unsigned* __restrict__ Ap,
               const unsigned* __restrict__ Bp,
               const float* __restrict__ bias,
               float* __restrict__ out) {
    const int lane  = threadIdx.x & 31;
    const int wave  = threadIdx.x >> 5;
    const int mtile = blockIdx.x;

    f32x8 acc0 = {}, acc1 = {}, acc2 = {}, acc3 = {};
    f32x8* acc[4] = {&acc0, &acc1, &acc2, &acc3};

    for (int kt = 0; kt < 16; ++kt) {
        // A fragment (shared by the 4 N-tiles): 4 coalesced b128 loads
        Frag ahi, alo;
        {
            const unsigned* at = Ap + (((size_t)mtile * 16 + kt) << 9);
#pragma unroll
            for (int c = 0; c < 2; ++c)
#pragma unroll
                for (int w = 0; w < 4; ++w) {
                    ahi.u[c * 4 + w] = at[c * 128 + lane * 4 + w];
                    alo.u[c * 4 + w] = at[(2 + c) * 128 + lane * 4 + w];
                }
        }
#pragma unroll
        for (int tdx = 0; tdx < 4; ++tdx) {
            const int ntile = wave * 4 + tdx;
            const unsigned* bt = Bp + (((size_t)kt * 16 + ntile) << 9);
            Frag bhi, blo;
#pragma unroll
            for (int c = 0; c < 2; ++c)
#pragma unroll
                for (int w = 0; w < 4; ++w) {
                    bhi.u[c * 4 + w] = bt[c * 128 + lane * 4 + w];
                    blo.u[c * 4 + w] = bt[(2 + c) * 128 + lane * 4 + w];
                }
            f32x8 c = *acc[tdx];
            c = __builtin_amdgcn_wmma_f32_16x16x32_bf16(false, ahi.v, false, bhi.v,
                                                        (short)0, c, false, false);
            c = __builtin_amdgcn_wmma_f32_16x16x32_bf16(false, ahi.v, false, blo.v,
                                                        (short)0, c, false, false);
            c = __builtin_amdgcn_wmma_f32_16x16x32_bf16(false, alo.v, false, bhi.v,
                                                        (short)0, c, false, false);
            *acc[tdx] = c;
        }
    }

    // Epilogue: C/D layout — VGPR v: lanes 0-15 -> (M=v, N=lane),
    // lanes 16-31 -> (M=v+8, N=lane-16).
    const int nloc  = lane & 15;
    const int mbase = mtile * 16 + (lane >> 4) * 8;
#pragma unroll
    for (int tdx = 0; tdx < 4; ++tdx) {
        const int n = (wave * 4 + tdx) * 16 + nloc;
        const float b = bias[n];
        const f32x8 c = *acc[tdx];
#pragma unroll
        for (int v = 0; v < 8; ++v) {
            float val = c[v] + b;
            val = val > 0.f ? val : 0.f;
            out[(size_t)(mbase + v) * 256 + n] = val;
        }
    }
}

// ---------------------------------------------------------------------------
// inv[n_id[i]] = i   (identity for this input, but honor it anyway)
// ---------------------------------------------------------------------------
__global__ void build_inv(const int* __restrict__ n_id, int* __restrict__ inv, int n) {
    int t = blockIdx.x * blockDim.x + threadIdx.x;
    if (t < n) inv[n_id[t]] = t;
}

// ---------------------------------------------------------------------------
// Link head: out[l] = blin + h2[ps].Wlin[0:256] + h2[pd].Wlin[256:512]
// ---------------------------------------------------------------------------
__global__ void link_head(const float* __restrict__ h2,
                          const int* __restrict__ link,
                          const int* __restrict__ inv,
                          const float* __restrict__ Wlin,
                          const float* __restrict__ blin,
                          float* __restrict__ out, int L) {
    int l = blockIdx.x * blockDim.x + threadIdx.x;
    if (l >= L) return;
    int ps = inv[link[2 * l + 0]];
    int pd = inv[link[2 * l + 1]];
    const float4* hs = (const float4*)(h2 + (size_t)ps * 256);
    const float4* hd = (const float4*)(h2 + (size_t)pd * 256);
    const float4* w0 = (const float4*)(Wlin);
    const float4* w1 = (const float4*)(Wlin + 256);
    float s = blin[0];
    for (int i = 0; i < 64; ++i) {
        float4 a = hs[i], wa = w0[i];
        s += a.x * wa.x + a.y * wa.y + a.z * wa.z + a.w * wa.w;
        float4 b = hd[i], wb = w1[i];
        s += b.x * wb.x + b.y * wb.y + b.z * wb.z + b.w * wb.w;
    }
    out[l] = s;
}

// ---------------------------------------------------------------------------
// Host launcher. Workspace layout (bytes), total needed ~= 124.1 MB:
//   [0,            81,920,000)  A1 packed (2500*16 tiles * 2KB)   -- dead after gemm1,
//                               then recycled for ALL layer-2 temporaries:
//        [0,          4,096,000)  agg2
//        [4,096,000,  4,112,000)  cnt2
//        [4,112,384, 12,304,384)  A2 packed (250*16 tiles * 2KB)
//        [12,304,384,16,400,384)  h2
//        [16,400,384,18,000,384)  inv (int32, N0)
//   [81,920,000, 122,880,000)  agg1 -> recycled as h1 after finalize1
//   [122,880,000,123,040,000)  cnt1
//   [123,040,000,123,564,288)  B1 packed
//   [123,564,288,124,088,576)  B2 packed
// ---------------------------------------------------------------------------
extern "C" void kernel_launch(void* const* d_in, const int* in_sizes, int n_in,
                              void* d_out, int out_size, void* d_ws, size_t ws_size,
                              hipStream_t stream) {
    const float* x    = (const float*)d_in[0];
    const int*   src1 = (const int*)d_in[1];
    const int*   dst1 = (const int*)d_in[2];
    const int*   src2 = (const int*)d_in[3];
    const int*   dst2 = (const int*)d_in[4];
    const int*   link = (const int*)d_in[7];
    const int*   n_id = (const int*)d_in[8];
    const float* Wl1  = (const float*)d_in[9];
    const float* bl1  = (const float*)d_in[10];
    const float* Wr1  = (const float*)d_in[11];
    const float* Wl2  = (const float*)d_in[12];
    const float* bl2  = (const float*)d_in[13];
    const float* Wr2  = (const float*)d_in[14];
    const float* Wlin = (const float*)d_in[15];
    const float* blin = (const float*)d_in[16];
    float*       out  = (float*)d_out;

    const int E1 = in_sizes[1];          // 400000
    const int E2 = in_sizes[3];          // 40000
    const int L  = in_sizes[7] / 2;      // 4096
    const int N0 = in_sizes[8];          // 400000
    const int N1 = 40000, N2 = 4000;     // reference constants

    char* w = (char*)d_ws;
    unsigned* A1   = (unsigned*)(w + 0);
    float*    agg2 = (float*)   (w + 0);
    float*    cnt2 = (float*)   (w + 4096000);
    unsigned* A2   = (unsigned*)(w + 4112384);
    float*    h2   = (float*)   (w + 12304384);
    int*      inv  = (int*)     (w + 16400384);
    float*    agg1 = (float*)   (w + 81920000);
    float*    h1   = agg1;               // recycled after finalize1
    float*    cnt1 = (float*)   (w + 122880000);
    unsigned* B1   = (unsigned*)(w + 123040000);
    unsigned* B2   = (unsigned*)(w + 123564288);

    // ---- layer 1 ----
    {   // zero agg1 + cnt1 (contiguous): 41,120,000 B = 2,570,000 float4
        const int n4 = 2570000;
        zero_f4<<<(n4 + 255) / 256, 256, 0, stream>>>((float4*)agg1, n4);
    }
    scatter_add<<<(E1 * 64 + 255) / 256, 256, 0, stream>>>(x, src1, dst1, agg1, cnt1, E1);
    finalize_pack_A<<<(N1 * 256 + 255) / 256, 256, 0, stream>>>(agg1, cnt1, x, A1, N1);
    pack_B<<<256, 256, 0, stream>>>(Wl1, Wr1, B1);
    gemm_bias_relu<<<N1 / 16, 128, 0, stream>>>(A1, B1, bl1, h1);

    // ---- layer 2 (A1 region recycled) ----
    {   // zero agg2 + cnt2 (contiguous): 4,112,000 B = 257,000 float4
        const int n4 = 257000;
        zero_f4<<<(n4 + 255) / 256, 256, 0, stream>>>((float4*)agg2, n4);
    }
    scatter_add<<<(E2 * 64 + 255) / 256, 256, 0, stream>>>(h1, src2, dst2, agg2, cnt2, E2);
    finalize_pack_A<<<(N2 * 256 + 255) / 256, 256, 0, stream>>>(agg2, cnt2, h1, A2, N2);
    pack_B<<<256, 256, 0, stream>>>(Wl2, Wr2, B2);
    gemm_bias_relu<<<N2 / 16, 128, 0, stream>>>(A2, B2, bl2, h2);

    // ---- link prediction head ----
    build_inv<<<(N0 + 255) / 256, 256, 0, stream>>>(n_id, inv, N0);
    link_head<<<(L + 255) / 256, 256, 0, stream>>>(h2, link, inv, Wlin, blin, out, L);
}
